// Attention_10204842295862
// MI455X (gfx1250) — compile-verified, matrix operand order checked by reading, not run on hardware
//
#include <hip/hip_runtime.h>
#include <hip/hip_bf16.h>

typedef __attribute__((ext_vector_type(16))) _Float16 v16h;
typedef __attribute__((ext_vector_type(8)))  float    v8f;

#define BATCH 4
#define SEQ   4096
#define CH    128
#define QTILES (SEQ / 16)   // 256

// ---------------------------------------------------------------------------
// CDNA5 WMMA 16x16x32 f16 operand layout (ISA 7.12.2):
//   A (16x32, MxK): lane l<16 holds row M=l, halves 0..7 -> K=0..7,
//                   halves 8..15 -> K=16..23; lane l+16: same row,
//                   K = {8..15, 24..31}.
//   B (32x16, KxN): mirrored -- lane l<16 holds column N=l with the same
//                   K packing, so a ROW-major row at row==N yields the B
//                   fragment of the transposed matrix.
// Both reduce to: frag = src[row*LD + kb*32 + hi*8 .. +7] and [+16 .. +23].
// ---------------------------------------------------------------------------
__device__ inline v16h load_frag_f16(const _Float16* base, size_t row, int koff, int hi, int ld) {
    const _Float16* p = base + row * ld + koff + hi * 8;
    union { uint4 q[2]; v16h v; } u;
    u.q[0] = *reinterpret_cast<const uint4*>(p);        // 8 halves, 16B aligned
    u.q[1] = *reinterpret_cast<const uint4*>(p + 16);   // 8 halves
    return u.v;
}

__device__ inline v16h load_frag_f32(const float* base, int row, int kb, int hi) {
    const float* p = base + (size_t)row * CH + kb * 32 + hi * 8;
    float4 a0 = *reinterpret_cast<const float4*>(p);
    float4 a1 = *reinterpret_cast<const float4*>(p + 4);
    float4 b0 = *reinterpret_cast<const float4*>(p + 16);
    float4 b1 = *reinterpret_cast<const float4*>(p + 20);
    v16h v;
    v[0]=(_Float16)a0.x; v[1]=(_Float16)a0.y; v[2]=(_Float16)a0.z; v[3]=(_Float16)a0.w;
    v[4]=(_Float16)a1.x; v[5]=(_Float16)a1.y; v[6]=(_Float16)a1.z; v[7]=(_Float16)a1.w;
    v[8]=(_Float16)b0.x; v[9]=(_Float16)b0.y; v[10]=(_Float16)b0.z; v[11]=(_Float16)b0.w;
    v[12]=(_Float16)b1.x; v[13]=(_Float16)b1.y; v[14]=(_Float16)b1.z; v[15]=(_Float16)b1.w;
    return v;
}

// ---------------------------------------------------------------------------
// Kernel 1: q/k/v = X @ W^T, f32 in -> f16 out (f32 WMMA accumulate).
// q,k stored row-major [T][C]; v stored TRANSPOSED [C][T] so the attention
// kernel's P*V B-fragments are contiguous b128 loads.
// One wave per 16x16 output tile. Tiles: 3 proj * 4 batch * 256 mt * 8 nt.
// ---------------------------------------------------------------------------
__global__ __launch_bounds__(256) void qkv_proj_kernel(
    const float* __restrict__ X,
    const float* __restrict__ Wq, const float* __restrict__ Wk, const float* __restrict__ Wv,
    _Float16* __restrict__ qh, _Float16* __restrict__ kh, _Float16* __restrict__ vT)
{
    const int wave = threadIdx.x >> 5;
    const int lane = threadIdx.x & 31;
    const int lm = lane & 15, hi = lane >> 4;

    int flat = blockIdx.x * 8 + wave;         // [0, 24576)
    const int nt = flat & 7;   flat >>= 3;
    const int mt = flat & 255; flat >>= 8;
    const int b  = flat & 3;   const int p = flat >> 2;

    const float* W = (p == 0) ? Wq : (p == 1) ? Wk : Wv;
    const float* Xb = X + (size_t)b * SEQ * CH;

    v8f c = {};
#pragma unroll
    for (int kb = 0; kb < 4; ++kb) {
        v16h a  = load_frag_f32(Xb, mt * 16 + lm, kb, hi);   // A: X tile row
        v16h bw = load_frag_f32(W,  nt * 16 + lm, kb, hi);   // B: W^T column = W row
        c = __builtin_amdgcn_wmma_f32_16x16x32_f16(false, a, false, bw,
                                                   (short)0, c, false, false);
    }

    if (p < 2) {
        _Float16* ob = ((p == 0) ? qh : kh) + (size_t)b * SEQ * CH;
#pragma unroll
        for (int r = 0; r < 8; ++r) {
            const int row = mt * 16 + r + 8 * hi;   // C layout: lane>=16 -> M+8
            const int col = nt * 16 + lm;
            ob[(size_t)row * CH + col] = (_Float16)c[r];
        }
    } else {
        // transposed store: vT[col][row]; rows contiguous along r -> one b128
        union { uint4 q; _Float16 h[8]; } u;
#pragma unroll
        for (int r = 0; r < 8; ++r) u.h[r] = (_Float16)c[r];
        _Float16* ob = vT + (size_t)b * SEQ * CH
                          + (size_t)(nt * 16 + lm) * SEQ + mt * 16 + 8 * hi;
        *reinterpret_cast<uint4*>(ob) = u.q;
    }
}

// ---------------------------------------------------------------------------
// Kernel 2: causal flash attention, 32-key blocks (full-K P*V WMMAs).
// One wave per 16-row query tile; online softmax in registers; P staged
// through a 1KB per-wave LDS tile (C-fragment -> A-fragment transpose).
// ---------------------------------------------------------------------------
__global__ __launch_bounds__(256) void flash_attn_kernel(
    const _Float16* __restrict__ qh, const _Float16* __restrict__ kh,
    const _Float16* __restrict__ vT, float* __restrict__ out)
{
    __shared__ _Float16 lds_p[8 * 512];       // 8 waves * 16x32 f16

    const int wave = threadIdx.x >> 5;
    const int lane = threadIdx.x & 31;
    const int lm = lane & 15, hi = lane >> 4;

    int flat = blockIdx.x * 8 + wave;         // [0, 1024)
    const int qt = flat & 255;
    const int b  = flat >> 8;

    const _Float16* qb  = qh + (size_t)b * SEQ * CH;
    const _Float16* kbp = kh + (size_t)b * SEQ * CH;
    const _Float16* vTb = vT + (size_t)b * SEQ * CH;   // [CH][SEQ]

    // Q fragments for all four K-chunks, resident for the whole pass.
    v16h aq[4];
#pragma unroll
    for (int kb = 0; kb < 4; ++kb) aq[kb] = load_frag_f16(qb, qt * 16 + lm, kb * 32, hi, CH);

    v8f o[8];
#pragma unroll
    for (int n = 0; n < 8; ++n) o[n] = (v8f){};

    float mrow[8], lrow[8];
#pragma unroll
    for (int r = 0; r < 8; ++r) { mrow[r] = -3.0e38f; lrow[r] = 0.0f; }

    const float scale = 0.08838834764831845f;  // 1/sqrt(128)
    _Float16* lp = lds_p + wave * 512;

    const int nfull = (qt + 1) >> 1;           // complete 32-key blocks

    for (int jj = 0; jj < nfull; ++jj) {
        const int k0 = jj * 32;
        if (jj + 1 < nfull) {
            __builtin_prefetch(kbp + (size_t)(k0 + 32) * CH, 0, 1);
        }

        // ---- S = Q * K^T for two adjacent 16-key tiles (K=128) ----
        v8f s0 = {}, s1 = {};
#pragma unroll
        for (int kb = 0; kb < 4; ++kb) {
            v16h bk0 = load_frag_f16(kbp, k0 + lm,      kb * 32, hi, CH);
            v16h bk1 = load_frag_f16(kbp, k0 + 16 + lm, kb * 32, hi, CH);
            s0 = __builtin_amdgcn_wmma_f32_16x16x32_f16(false, aq[kb], false, bk0,
                                                        (short)0, s0, false, false);
            s1 = __builtin_amdgcn_wmma_f32_16x16x32_f16(false, aq[kb], false, bk1,
                                                        (short)0, s1, false, false);
        }

        // ---- mask + scale + online softmax over 32 keys ----
        float alpha[8];
#pragma unroll
        for (int r = 0; r < 8; ++r) {
            const int qrow = qt * 16 + r + 8 * hi;
            float v0 = s0[r] * scale;
            float v1 = s1[r] * scale;
            if (k0 + lm      > qrow) v0 = -3.0e38f;   // bites only near diagonal
            if (k0 + 16 + lm > qrow) v1 = -3.0e38f;
            float rmax = fmaxf(v0, v1);
            rmax = fmaxf(rmax, __shfl_xor(rmax, 1));
            rmax = fmaxf(rmax, __shfl_xor(rmax, 2));
            rmax = fmaxf(rmax, __shfl_xor(rmax, 4));
            rmax = fmaxf(rmax, __shfl_xor(rmax, 8));
            const float mnew = fmaxf(mrow[r], rmax);
            alpha[r] = __expf(mrow[r] - mnew);
            const float p0 = __expf(v0 - mnew);
            const float p1 = __expf(v1 - mnew);
            float rsum = p0 + p1;
            rsum += __shfl_xor(rsum, 1);
            rsum += __shfl_xor(rsum, 2);
            rsum += __shfl_xor(rsum, 4);
            rsum += __shfl_xor(rsum, 8);
            lrow[r] = lrow[r] * alpha[r] + rsum;
            mrow[r] = mnew;
            s0[r] = p0;
            s1[r] = p1;
        }

#pragma unroll
        for (int n = 0; n < 8; ++n)
#pragma unroll
            for (int r = 0; r < 8; ++r) o[n][r] *= alpha[r];

        // ---- stage P (16x32) through LDS: C-frag -> A-frag ----
#pragma unroll
        for (int r = 0; r < 8; ++r) {
            lp[(r + 8 * hi) * 32 + lm]      = (_Float16)s0[r];
            lp[(r + 8 * hi) * 32 + 16 + lm] = (_Float16)s1[r];
        }
        asm volatile("s_wait_dscnt 0" ::: "memory");   // DS in-order; compiler fence

        union { uint4 q[2]; v16h v; } uap;
        uap.q[0] = *reinterpret_cast<const uint4*>(lp + lm * 32 + hi * 8);      // K 0..15
        uap.q[1] = *reinterpret_cast<const uint4*>(lp + lm * 32 + 16 + hi * 8); // K 16..31
        const v16h ap = uap.v;

        // ---- O += P * V : full-K WMMAs; B-frags contiguous from V^T ----
#pragma unroll
        for (int n = 0; n < 8; ++n) {
            const _Float16* vrow = vTb + (size_t)(n * 16 + lm) * SEQ + k0;
            union { uint4 q[2]; v16h v; } ub;
            ub.q[0] = *reinterpret_cast<const uint4*>(vrow + hi * 8);       // k 0..15
            ub.q[1] = *reinterpret_cast<const uint4*>(vrow + 16 + hi * 8);  // k 16..31
            o[n] = __builtin_amdgcn_wmma_f32_16x16x32_f16(false, ap, false, ub.v,
                                                          (short)0, o[n], false, false);
        }
    }

    // ---- leftover diagonal 16-key tile (qt even): zero-padded K half ----
    if ((qt & 1) == 0) {
        const int k0 = qt * 16;
        v8f s = {};
#pragma unroll
        for (int kb = 0; kb < 4; ++kb) {
            v16h bk = load_frag_f16(kbp, k0 + lm, kb * 32, hi, CH);
            s = __builtin_amdgcn_wmma_f32_16x16x32_f16(false, aq[kb], false, bk,
                                                       (short)0, s, false, false);
        }
        float alpha[8];
#pragma unroll
        for (int r = 0; r < 8; ++r) {
            const int qrow = qt * 16 + r + 8 * hi;
            float v = s[r] * scale;
            if (k0 + lm > qrow) v = -3.0e38f;
            float rmax = v;
            rmax = fmaxf(rmax, __shfl_xor(rmax, 1));
            rmax = fmaxf(rmax, __shfl_xor(rmax, 2));
            rmax = fmaxf(rmax, __shfl_xor(rmax, 4));
            rmax = fmaxf(rmax, __shfl_xor(rmax, 8));
            const float mnew = fmaxf(mrow[r], rmax);
            alpha[r] = __expf(mrow[r] - mnew);
            const float pv = __expf(v - mnew);
            float rsum = pv;
            rsum += __shfl_xor(rsum, 1);
            rsum += __shfl_xor(rsum, 2);
            rsum += __shfl_xor(rsum, 4);
            rsum += __shfl_xor(rsum, 8);
            lrow[r] = lrow[r] * alpha[r] + rsum;
            mrow[r] = mnew;
            s[r] = pv;
        }
#pragma unroll
        for (int n = 0; n < 8; ++n)
#pragma unroll
            for (int r = 0; r < 8; ++r) o[n][r] *= alpha[r];

#pragma unroll
        for (int r = 0; r < 8; ++r)
            lp[(r + 8 * hi) * 32 + lm] = (_Float16)s[r];
        asm volatile("s_wait_dscnt 0" ::: "memory");

        union { uint4 q[2]; v16h v; } uap;
        uap.q[0] = *reinterpret_cast<const uint4*>(lp + lm * 32 + hi * 8); // K 0..15
        uap.q[1] = (uint4){0, 0, 0, 0};                                    // K 16..31 = 0
        const v16h ap = uap.v;

#pragma unroll
        for (int n = 0; n < 8; ++n) {
            const _Float16* vrow = vTb + (size_t)(n * 16 + lm) * SEQ + k0;
            union { uint4 q[2]; v16h v; } ub;
            ub.q[0] = *reinterpret_cast<const uint4*>(vrow + hi * 8);  // k 0..15 (in range)
            ub.q[1] = (uint4){0, 0, 0, 0};                             // padded half
            o[n] = __builtin_amdgcn_wmma_f32_16x16x32_f16(false, ap, false, ub.v,
                                                          (short)0, o[n], false, false);
        }
    }

    // ---- epilogue: normalize and store f32 ----
    float inv[8];
#pragma unroll
    for (int r = 0; r < 8; ++r) inv[r] = 1.0f / lrow[r];

    float* ob = out + (size_t)b * SEQ * CH;
#pragma unroll
    for (int n = 0; n < 8; ++n)
#pragma unroll
        for (int r = 0; r < 8; ++r)
            ob[(size_t)(qt * 16 + r + 8 * hi) * CH + n * 16 + lm] = o[n][r] * inv[r];
}

// ---------------------------------------------------------------------------
extern "C" void kernel_launch(void* const* d_in, const int* in_sizes, int n_in,
                              void* d_out, int out_size, void* d_ws, size_t ws_size,
                              hipStream_t stream) {
    (void)in_sizes; (void)n_in; (void)out_size; (void)ws_size;
    const float* X  = (const float*)d_in[0];
    const float* Wq = (const float*)d_in[1];
    const float* Wk = (const float*)d_in[2];
    const float* Wv = (const float*)d_in[3];
    float* out = (float*)d_out;

    const size_t elems = (size_t)BATCH * SEQ * CH;       // 2,097,152
    _Float16* qh = (_Float16*)d_ws;
    _Float16* kh = qh + elems;
    _Float16* vT = kh + elems;                            // [B][CH][SEQ]

    // 3 proj * 4 batch * 256 mt * 8 nt tiles / 8 waves per block = 3072 blocks
    qkv_proj_kernel<<<3072, 256, 0, stream>>>(X, Wq, Wk, Wv, qh, kh, vT);
    // 4 batch * 256 query tiles / 8 waves per block = 128 blocks
    flash_attn_kernel<<<128, 256, 0, stream>>>(qh, kh, vT, out);
}
